// SentimentAnalysis_13829794693817
// MI455X (gfx1250) — compile-verified, matrix operand order checked by reading, not run on hardware
//
#include <hip/hip_runtime.h>
#include <math.h>

// ---------------- CDNA5 WMMA plumbing (wave32, 16x16x32 bf16 -> f32) -------
typedef __attribute__((ext_vector_type(16))) __bf16 bf16x16;
typedef __attribute__((ext_vector_type(8)))  float  f32x8;

union FragBF {
    bf16x16 v;
    uint4   q[2];
    unsigned short u[16];
};

__device__ inline f32x8 wmma_bf16f32(bf16x16 a, bf16x16 b, f32x8 c) {
#if defined(__gfx1250__)
    // 8 args: (neg_a, A, neg_b, B, c_mod, C, reuse_a, reuse_b)
    return __builtin_amdgcn_wmma_f32_16x16x32_bf16(false, a, false, b,
                                                   (short)0, c, false, false);
#else
    return c; // host pass / other targets: never executed on device
#endif
}

__device__ inline unsigned short f2bf(float f) {
    unsigned int u = __float_as_uint(f);
    u = (u + 0x7FFFu + ((u >> 16) & 1u)) >> 16;   // RNE
    return (unsigned short)u;
}

__device__ inline float sigm(float x) { return 1.0f / (1.0f + expf(-x)); }

// A fragment: 16x32 bf16 tile of a row-major matrix (ld = row stride, elements).
// ISA layout: lanes 0-15 hold M=lane (K = 0..7,16..23), lanes 16-31 hold
// M=lane-16 (K = 8..15,24..31): two contiguous 16B loads per lane.
__device__ inline bf16x16 load_frag_rowmajor(const unsigned short* base, int ld,
                                             int row0, int k0, int lane) {
    int r    = row0 + (lane & 15);
    int koff = (lane >> 4) << 3;
    const unsigned short* p = base + (size_t)r * ld + k0 + koff;
    FragBF f;
    f.q[0] = *(const uint4*)(p);
    f.q[1] = *(const uint4*)(p + 16);
    return f.v;
}

// B fragment from quad-packed weights: per (quad-of-4-n-tiles, k-tile) the 4
// sub-tiles' lane data are contiguous (4KB chunk); each lane reads 32B.
__device__ inline bf16x16 load_frag_quad(const unsigned short* Wp, int KT,
                                         int tq, int kt, int sub, int lane) {
    const unsigned short* p =
        Wp + (((((size_t)tq * KT + kt) * 4 + sub) * 32 + lane) << 4);
    FragBF f;
    f.q[0] = *(const uint4*)(p);
    f.q[1] = *(const uint4*)(p + 8);
    return f.v;
}

// ---------------- utility kernels ------------------------------------------
__global__ void zero_k(unsigned int* p, size_t nwords) {
    size_t i = (size_t)blockIdx.x * blockDim.x + threadIdx.x;
    if (i < nwords) p[i] = 0u;
}

// x [B=64][T=512][E=300] f32  ->  xb [T][B][KXP=320] bf16 (zero-padded K)
__global__ void xconv_k(const float* __restrict__ x, unsigned short* __restrict__ xb) {
    size_t idx = (size_t)blockIdx.x * blockDim.x + threadIdx.x;
    const size_t total = (size_t)512 * 64 * 320;
    if (idx >= total) return;
    int k = (int)(idx % 320);
    size_t r = idx / 320;
    int b = (int)(r % 64);
    int t = (int)(r / 64);
    float v = (k < 300) ? x[((size_t)b * 512 + t) * 300 + k] : 0.0f;
    xb[idx] = f2bf(v);
}

// W f32 [Ksrc][N] row-major -> quad-packed bf16 B-fragment layout (K pad Kpad).
// idx = ((((tq*KT + kt)*4 + sub)*32 + lane)*16 + e
__global__ void pack_w_k(const float* __restrict__ W, unsigned short* __restrict__ Wp,
                         int Ksrc, int Kpad, int N) {
    size_t idx = (size_t)blockIdx.x * blockDim.x + threadIdx.x;
    size_t total = (size_t)Kpad * N;
    if (idx >= total) return;
    int e = (int)(idx & 15);
    size_t rest = idx >> 4;
    int lane = (int)(rest & 31);
    rest >>= 5;
    int sub = (int)(rest & 3);
    rest >>= 2;
    int KT = Kpad >> 5;
    int kt = (int)(rest % KT);
    int tq = (int)(rest / KT);
    int koff = (lane >> 4) << 3;
    int k = kt * 32 + koff + (e < 8 ? e : e + 8);
    int n = (tq * 4 + sub) * 16 + (lane & 15);
    float v = (k < Ksrc) ? W[(size_t)k * N + n] : 0.0f;
    Wp[idx] = f2bf(v);
}

// ---------------- LSTM step: gate GEMM (WMMA) ------------------------------
// G[64][N] = A0[64][32*KT0] @ W0 + A1[64][32*KT1] @ W1 + bias
// Each wave: one 16-row M tile x four 16-col N tiles (4 accumulator chains).
// All 10 loads of a k-tile are issued before its 4 WMMAs (no cross-kt
// buffering -> no spills; 5 live fragments + 4 accumulators ~ 72 VGPRs).
struct GemmDir {
    const unsigned short* A0; int lda0; int KT0; const unsigned short* W0;
    const unsigned short* A1; int lda1; int KT1; const unsigned short* W1;
    const float* bias; float* G;
};

__device__ inline void gemm_seg(const unsigned short* A, int lda, int KT,
                                const unsigned short* Wp, int tm, int tq, int lane,
                                f32x8 (&acc)[4]) {
    for (int kt = 0; kt < KT; ++kt) {
        if (kt + 1 < KT)  // one prefetch covers the next contiguous 4KB chunk
            __builtin_prefetch(
                Wp + ((size_t)(tq * KT + kt + 1) * 2048 + (size_t)lane * 64),
                0, 3);
        bf16x16 a  = load_frag_rowmajor(A, lda, tm * 16, kt * 32, lane);
        bf16x16 b0 = load_frag_quad(Wp, KT, tq, kt, 0, lane);
        bf16x16 b1 = load_frag_quad(Wp, KT, tq, kt, 1, lane);
        bf16x16 b2 = load_frag_quad(Wp, KT, tq, kt, 2, lane);
        bf16x16 b3 = load_frag_quad(Wp, KT, tq, kt, 3, lane);
        acc[0] = wmma_bf16f32(a, b0, acc[0]);
        acc[1] = wmma_bf16f32(a, b1, acc[1]);
        acc[2] = wmma_bf16f32(a, b2, acc[2]);
        acc[3] = wmma_bf16f32(a, b3, acc[3]);
    }
}

__global__ void gate_gemm_k(GemmDir d0, GemmDir d1, int N) {
    GemmDir d = (blockIdx.y == 0) ? d0 : d1;
    int lane = threadIdx.x & 31;
    int wave = threadIdx.x >> 5;
    int wq = blockIdx.x * (blockDim.x >> 5) + wave;  // quad-tile id
    int ntq = N >> 6;                                // quads of 4 n-tiles
    int tm = wq / ntq;
    int tq = wq - tm * ntq;
    if (tm >= 4) return;
    f32x8 acc[4];
#pragma unroll
    for (int s = 0; s < 4; ++s) acc[s] = (f32x8){0.f,0.f,0.f,0.f,0.f,0.f,0.f,0.f};

    gemm_seg(d.A0, d.lda0, d.KT0, d.W0, tm, tq, lane, acc);
    gemm_seg(d.A1, d.lda1, d.KT1, d.W1, tm, tq, lane, acc);

    int mhi = (lane >> 4) << 3;  // C/D layout: VGPR r -> M = r + (lane<16?0:8)
#pragma unroll
    for (int sub = 0; sub < 4; ++sub) {
        int n = (tq * 4 + sub) * 16 + (lane & 15);
        float bv = d.bias[n];
        float* Gp = d.G + (size_t)(tm * 16 + mhi) * N + n;
#pragma unroll
        for (int r = 0; r < 8; ++r) Gp[(size_t)r * N] = acc[sub][r] + bv;
    }
}

// ---------------- LSTM step: element-wise gates ----------------------------
struct PointDir {
    const float* G;      // [64][4*Hh], gate order i,f,g,o
    float* c;            // [64][Hh] cell state (in/out)
    unsigned short* hb;  // bf16 h out (row stride hb_ld)
    int hb_ld;
    float* hf;           // optional f32 h out (row stride 1024), nullptr if unused
};

__global__ void lstm_point_k(PointDir p0, PointDir p1, int Hh) {
    PointDir p = (blockIdx.y == 0) ? p0 : p1;
    int idx = blockIdx.x * blockDim.x + threadIdx.x;
    if (idx >= 64 * Hh) return;
    int b = idx / Hh;
    int j = idx - b * Hh;
    const float* g = p.G + (size_t)b * 4 * Hh;
    float iv = sigm(g[j]);
    float fv = sigm(g[Hh + j]);
    float gv = tanhf(g[2 * Hh + j]);
    float ov = sigm(g[3 * Hh + j]);
    float cc = fv * p.c[idx] + iv * gv;
    p.c[idx] = cc;
    float h = ov * tanhf(cc);
    p.hb[(size_t)b * p.hb_ld + j] = f2bf(h);
    if (p.hf) p.hf[(size_t)b * 1024 + j] = h;
}

// ---------------- attention + head -----------------------------------------
// scores[t][b] = sum_h tanh(hs[t][b][h]) * w[h]   (one wave per (t,b))
__global__ void attn_score_k(const float* __restrict__ hs, const float* __restrict__ w,
                             float* __restrict__ scores) {
    int wid = blockIdx.x * 8 + (threadIdx.x >> 5);
    int lane = threadIdx.x & 31;
    int t = wid >> 6, b = wid & 63;
    const float* h = hs + ((size_t)t * 64 + b) * 1024;
    float s = 0.f;
    for (int k = lane; k < 1024; k += 32) s += tanhf(h[k]) * w[k];
    for (int o = 16; o; o >>= 1) s += __shfl_down(s, o, 32);
    if (lane == 0) scores[t * 64 + b] = s;
}

// softmax over T per batch column; writes alpha directly as [B][T] (= alpha.T)
__global__ void softmax_t_k(const float* __restrict__ scores, float* __restrict__ alphaBT) {
    __shared__ float sm[256];
    int b = blockIdx.x, tid = threadIdx.x;
    float mx = -3.4e38f;
    for (int t = tid; t < 512; t += 256) mx = fmaxf(mx, scores[t * 64 + b]);
    sm[tid] = mx; __syncthreads();
    for (int s = 128; s; s >>= 1) { if (tid < s) sm[tid] = fmaxf(sm[tid], sm[tid + s]); __syncthreads(); }
    mx = sm[0]; __syncthreads();
    float sum = 0.f;
    for (int t = tid; t < 512; t += 256) sum += expf(scores[t * 64 + b] - mx);
    sm[tid] = sum; __syncthreads();
    for (int s = 128; s; s >>= 1) { if (tid < s) sm[tid] += sm[tid + s]; __syncthreads(); }
    float inv = 1.f / sm[0];
    for (int t = tid; t < 512; t += 256)
        alphaBT[(size_t)b * 512 + t] = expf(scores[t * 64 + b] - mx) * inv;
}

// ctx[b][j] = sum_t hs[t][b][j] * alpha[b][t]
__global__ void ctx_k(const float* __restrict__ hs, const float* __restrict__ alphaBT,
                      float* __restrict__ ctx) {
    int idx = blockIdx.x * blockDim.x + threadIdx.x;   // 64*1024
    int b = idx >> 10, j = idx & 1023;
    float acc = 0.f;
    for (int t = 0; t < 512; ++t)
        acc += hs[((size_t)t * 64 + b) * 1024 + j] * alphaBT[(size_t)b * 512 + t];
    ctx[idx] = acc;
}

// h = relu(ctx @ fc_W + fc_b)
__global__ void fc_k(const float* __restrict__ ctx, const float* __restrict__ W,
                     const float* __restrict__ bias, float* __restrict__ out) {
    int idx = blockIdx.x * blockDim.x + threadIdx.x;   // 64*1024
    int b = idx >> 10, n = idx & 1023;
    float acc = bias[n];
    for (int k = 0; k < 1024; ++k) acc += ctx[(size_t)b * 1024 + k] * W[(size_t)k * 1024 + n];
    out[idx] = fmaxf(acc, 0.f);
}

// out[b][c] = h[b] . lin_W[:,c] + lin_b[c]   (one wave per (b,c))
__global__ void lin_k(const float* __restrict__ h, const float* __restrict__ W,
                      const float* __restrict__ bias, float* __restrict__ out) {
    int wid = blockIdx.x * 8 + (threadIdx.x >> 5);
    if (wid >= 128) return;
    int lane = threadIdx.x & 31;
    int b = wid >> 1, c = wid & 1;
    float s = 0.f;
    for (int k = lane; k < 1024; k += 32) s += h[(size_t)b * 1024 + k] * W[(size_t)k * 2 + c];
    for (int o = 16; o; o >>= 1) s += __shfl_down(s, o, 32);
    if (lane == 0) out[b * 2 + c] = s + bias[c];
}

// ---------------- host orchestration ---------------------------------------
extern "C" void kernel_launch(void* const* d_in, const int* in_sizes, int n_in,
                              void* d_out, int out_size, void* d_ws, size_t ws_size,
                              hipStream_t stream) {
    (void)in_sizes; (void)n_in; (void)out_size; (void)ws_size;
    const int B = 64, T = 512;

    const float* x      = (const float*)d_in[0];
    const float* Wx_f   = (const float*)d_in[1];
    const float* Wh_f   = (const float*)d_in[2];
    const float* b_f    = (const float*)d_in[3];
    const float* Wx_b   = (const float*)d_in[4];
    const float* Wh_b   = (const float*)d_in[5];
    const float* b_b    = (const float*)d_in[6];
    const float* Wx_s   = (const float*)d_in[7];
    const float* Wh_s   = (const float*)d_in[8];
    const float* b_s    = (const float*)d_in[9];
    const float* attn_w = (const float*)d_in[10];
    const float* fc_W   = (const float*)d_in[11];
    const float* fc_b   = (const float*)d_in[12];
    const float* lin_W  = (const float*)d_in[13];
    const float* lin_b  = (const float*)d_in[14];
    float* out = (float*)d_out;

    // workspace carve-up (deterministic)
    char* W = (char*)d_ws;
    size_t off = 0;
    auto take = [&](size_t bytes) -> char* {
        char* p = W + off;
        off = (off + bytes + 255) & ~(size_t)255;
        return p;
    };
    // zero-initialized region first
    unsigned short* zbuf = (unsigned short*)take((size_t)B * 1024 * 2); // zero h (bf16)
    float* c_f = (float*)take((size_t)B * 512 * 4);
    float* c_b = (float*)take((size_t)B * 512 * 4);
    float* c_s = (float*)take((size_t)B * 1024 * 4);
    size_t zero_bytes = off;
    // persistent converted operands
    unsigned short* xb    = (unsigned short*)take((size_t)T * B * 320 * 2);
    unsigned short* WxfP  = (unsigned short*)take((size_t)320 * 2048 * 2);
    unsigned short* WhfP  = (unsigned short*)take((size_t)512 * 2048 * 2);
    unsigned short* WxbP  = (unsigned short*)take((size_t)320 * 2048 * 2);
    unsigned short* WhbP  = (unsigned short*)take((size_t)512 * 2048 * 2);
    unsigned short* WxsP  = (unsigned short*)take((size_t)1024 * 4096 * 2);
    unsigned short* WhsP  = (unsigned short*)take((size_t)1024 * 4096 * 2);
    unsigned short* hcat  = (unsigned short*)take((size_t)T * B * 1024 * 2); // layer1 outputs (bf16)
    unsigned short* hsbuf = (unsigned short*)take((size_t)2 * B * 1024 * 2); // layer2 recurrent (dbl buf)
    float* hs     = (float*)take((size_t)T * B * 1024 * 4);                  // layer2 outputs (f32)
    float* G_f    = (float*)take((size_t)B * 2048 * 4);
    float* G_b    = (float*)take((size_t)B * 2048 * 4);
    float* G_s    = (float*)take((size_t)B * 4096 * 4);
    float* scores = (float*)take((size_t)T * B * 4);
    float* ctx    = (float*)take((size_t)B * 1024 * 4);
    float* fch    = (float*)take((size_t)B * 1024 * 4);

    // 0) zero the state region
    {
        size_t nw = zero_bytes / 4;
        zero_k<<<(unsigned)((nw + 255) / 256), 256, 0, stream>>>((unsigned int*)d_ws, nw);
    }
    // 1) convert x and pack weights into WMMA-fragment layout (L2-resident)
    {
        size_t n = (size_t)T * B * 320;
        xconv_k<<<(unsigned)((n + 255) / 256), 256, 0, stream>>>(x, xb);
        auto pk = [&](const float* src, unsigned short* dst, int Ks, int Kp, int N) {
            size_t tot = (size_t)Kp * N;
            pack_w_k<<<(unsigned)((tot + 255) / 256), 256, 0, stream>>>(src, dst, Ks, Kp, N);
        };
        pk(Wx_f, WxfP, 300, 320, 2048);
        pk(Wh_f, WhfP, 512, 512, 2048);
        pk(Wx_b, WxbP, 300, 320, 2048);
        pk(Wh_b, WhbP, 512, 512, 2048);
        pk(Wx_s, WxsP, 1024, 1024, 4096);
        pk(Wh_s, WhsP, 1024, 1024, 4096);
    }

    // 2) layer 1: fwd + bwd LSTM chains fused per step via blockIdx.y
    for (int t = 0; t < T; ++t) {
        int tb = T - 1 - t;
        GemmDir df, db;
        df.A0 = xb + (size_t)t * B * 320;  df.lda0 = 320;  df.KT0 = 10; df.W0 = WxfP;
        df.A1 = (t == 0) ? zbuf : hcat + (size_t)(t - 1) * B * 1024;
        df.lda1 = 1024; df.KT1 = 16; df.W1 = WhfP; df.bias = b_f; df.G = G_f;
        db.A0 = xb + (size_t)tb * B * 320; db.lda0 = 320;  db.KT0 = 10; db.W0 = WxbP;
        db.A1 = (t == 0) ? zbuf : hcat + (size_t)(tb + 1) * B * 1024 + 512;
        db.lda1 = 1024; db.KT1 = 16; db.W1 = WhbP; db.bias = b_b; db.G = G_b;
        gate_gemm_k<<<dim3(16, 2), 256, 0, stream>>>(df, db, 2048);

        PointDir pf{G_f, c_f, hcat + (size_t)t * B * 1024, 1024, nullptr};
        PointDir pb{G_b, c_b, hcat + (size_t)tb * B * 1024 + 512, 1024, nullptr};
        lstm_point_k<<<dim3(128, 2), 256, 0, stream>>>(pf, pb, 512);
    }

    // 3) layer 2: LSTM over hcat
    for (int t = 0; t < T; ++t) {
        GemmDir ds;
        ds.A0 = hcat + (size_t)t * B * 1024; ds.lda0 = 1024; ds.KT0 = 32; ds.W0 = WxsP;
        ds.A1 = (t == 0) ? zbuf : hsbuf + (size_t)((t - 1) & 1) * B * 1024;
        ds.lda1 = 1024; ds.KT1 = 32; ds.W1 = WhsP; ds.bias = b_s; ds.G = G_s;
        gate_gemm_k<<<dim3(32, 1), 256, 0, stream>>>(ds, ds, 4096);

        PointDir ps{G_s, c_s, hsbuf + (size_t)(t & 1) * B * 1024, 1024,
                    hs + (size_t)t * B * 1024};
        lstm_point_k<<<dim3(256, 1), 256, 0, stream>>>(ps, ps, 1024);
    }

    // 4) attention pooling + head; alpha.T written straight into d_out[128:]
    attn_score_k<<<4096, 256, 0, stream>>>(hs, attn_w, scores);
    softmax_t_k<<<64, 256, 0, stream>>>(scores, out + 128);
    ctx_k<<<256, 256, 0, stream>>>(hs, out + 128, ctx);
    fc_k<<<256, 256, 0, stream>>>(ctx, fc_W, fc_b, fch);
    lin_k<<<16, 256, 0, stream>>>(fch, lin_W, lin_b, out);
}
